// StatsColorBranch_26328149524989
// MI455X (gfx1250) — compile-verified, matrix operand order checked by reading, not run on hardware
//
#include <hip/hip_runtime.h>
#include <hip/hip_bf16.h>
#include <math.h>

typedef __attribute__((ext_vector_type(16))) _Float16 v16h;
typedef __attribute__((ext_vector_type(8)))  float    v8f;
typedef __attribute__((ext_vector_type(2)))  float    v2f;
typedef __attribute__((ext_vector_type(4)))  unsigned int u32x4;
typedef __attribute__((ext_vector_type(8)))  int      i32x8;
typedef __attribute__((ext_vector_type(4)))  int      i32x4;

#define HW 65536      // 256*256
#define FVW 40        // 39 features padded to 40 for WMMA K-tiling
#define NSL 8         // row-slices per image for stats partials
#define NPART 44      // 32 float sums + 12 hist bins per partial

// ---------------------------------------------------------------- zero
__global__ void k_zero(float* p, int n) {
  int i = blockIdx.x * 256 + threadIdx.x;
  if (i < n) p[i] = 0.f;
}

// ---------------------------------------------------------------- per-(image,slice) partial stats
// 1024 blocks x 256 threads: each block reduces 32 rows of one image.
__global__ void k_stats_part(const float* __restrict__ x, float* __restrict__ parts) {
  const int blk = blockIdx.x;
  const int b = blk >> 3, sl = blk & 7;
  const int tid = threadIdx.x;          // 256 threads
  const float* img = x + (size_t)b * 3 * HW;
  const int p0 = sl * (HW / NSL);       // 8192 pixels per slice

  __shared__ int   hist[12];
  __shared__ float red[32 * 256];
  if (tid < 12) hist[tid] = 0;
  __syncthreads();

  float s1[3] = {0,0,0}, s2[3] = {0,0,0}, s3[3] = {0,0,0}, s4[3] = {0,0,0};
  float sy=0, scb=0, scr=0, syy=0, sycb=0, sycr=0, scbcb=0, scbcr=0, scrcr=0;
  float ss=0, sss=0;
  float sg[3] = {0,0,0}, sgg[3] = {0,0,0}, sgx[3] = {0,0,0};

  for (int q = tid; q < HW / NSL; q += 256) {
    const int p = p0 + q;
    const int h = p >> 8, w = p & 255;
    __builtin_prefetch(&img[p + 256], 0, 3);
    const float r  = img[p];
    const float g  = img[HW + p];
    const float bl = img[2 * HW + p];
    const float v3[3] = {r, g, bl};
#pragma unroll
    for (int c = 0; c < 3; c++) {
      float v = v3[c], v2 = v * v;
      s1[c] += v; s2[c] += v2; s3[c] += v2 * v; s4[c] += v2 * v2;
    }
    const float y  =  0.299f    * r + 0.587f    * g + 0.114f    * bl;
    const float cb = -0.168736f * r - 0.331264f * g + 0.5f      * bl + 0.5f;
    const float cr =  0.5f      * r - 0.418688f * g - 0.081312f * bl + 0.5f;
    sy += y; scb += cb; scr += cr;
    syy += y*y; sycb += y*cb; sycr += y*cr;
    scbcb += cb*cb; scbcr += cb*cr; scrcr += cr*cr;

    const float maxc = fmaxf(r, fmaxf(g, bl));
    const float minc = fminf(r, fminf(g, bl));
    const float s = (maxc - minc) / (maxc + 1e-6f);
    ss += s; sss += s * s;
    const float d = maxc - minc + 1e-6f;
    const float rc = (maxc - r) / d, gc = (maxc - g) / d, bc = (maxc - bl) / d;
    float hh;                 // priority b > g > r (reference: later writes win)
    if (maxc == bl)     hh = 4.f + (gc - rc);
    else if (maxc == g) hh = 2.f + (rc - bc);
    else                hh = bc - gc;
    float h6 = hh * (1.f / 6.f);
    h6 = h6 - floorf(h6);                        // python %1.0
    int idx = (int)floorf(h6 * 12.f);
    idx = idx < 0 ? 0 : (idx > 11 ? 11 : idx);
    atomicAdd(&hist[idx], 1);

    float gm[3];
#pragma unroll
    for (int c = 0; c < 3; c++) {      // jnp.gradient: central diff, one-sided edges
      const float* pl = img + c * HW;
      const float cv = v3[c];
      const float up = (h > 0)   ? pl[p - 256] : 0.f;
      const float dn = (h < 255) ? pl[p + 256] : 0.f;
      const float lt = (w > 0)   ? pl[p - 1]   : 0.f;
      const float rt = (w < 255) ? pl[p + 1]   : 0.f;
      const float gx = (h == 0) ? (dn - cv) : (h == 255) ? (cv - up) : 0.5f * (dn - up);
      const float gy = (w == 0) ? (rt - cv) : (w == 255) ? (cv - lt) : 0.5f * (rt - lt);
      gm[c] = sqrtf(gx * gx + gy * gy);
    }
    sg[0]+=gm[0]; sg[1]+=gm[1]; sg[2]+=gm[2];
    sgg[0]+=gm[0]*gm[0]; sgg[1]+=gm[1]*gm[1]; sgg[2]+=gm[2]*gm[2];
    sgx[0]+=gm[0]*gm[1]; sgx[1]+=gm[0]*gm[2]; sgx[2]+=gm[1]*gm[2];
  }

#pragma unroll
  for (int c = 0; c < 3; c++) {
    red[(c*4+0)*256+tid]=s1[c]; red[(c*4+1)*256+tid]=s2[c];
    red[(c*4+2)*256+tid]=s3[c]; red[(c*4+3)*256+tid]=s4[c];
  }
  red[12*256+tid]=sy;   red[13*256+tid]=scb;   red[14*256+tid]=scr;
  red[15*256+tid]=syy;  red[16*256+tid]=sycb;  red[17*256+tid]=sycr;
  red[18*256+tid]=scbcb;red[19*256+tid]=scbcr; red[20*256+tid]=scrcr;
  red[21*256+tid]=ss;   red[22*256+tid]=sss;
#pragma unroll
  for (int c = 0; c < 3; c++) {
    red[(23+c)*256+tid]=sg[c]; red[(26+c)*256+tid]=sgg[c]; red[(29+c)*256+tid]=sgx[c];
  }
  __syncthreads();
  float* pp = parts + (size_t)blk * NPART;
  if (tid < 32) {
    float s = 0;
    for (int i = 0; i < 256; i++) s += red[tid * 256 + i];
    pp[tid] = s;
  }
  if (tid >= 32 && tid < 44) pp[tid] = (float)hist[tid - 32];
}

// ---------------------------------------------------------------- per-image finalize
// cols 0..11 moments, 12..17 ycbcr cov, 18/19 cb/cr mean, 35 hue-entropy,
// 36/37 sat mean/std, 38 grad-corr
__global__ void k_stats_final(const float* __restrict__ parts, float* __restrict__ fv) {
  const int b = blockIdx.x;
  const int tid = threadIdx.x;          // 64 threads
  __shared__ float tot[NPART];
  if (tid < NPART) {
    float s = 0;
    for (int sl = 0; sl < NSL; sl++) s += parts[(size_t)(b * NSL + sl) * NPART + tid];
    tot[tid] = s;
  }
  __syncthreads();
  if (tid != 0) return;

  const float N = 65536.f;
  float* f = fv + b * FVW;
#pragma unroll
  for (int c = 0; c < 3; c++) {
    const float S1=tot[c*4], S2=tot[c*4+1], S3=tot[c*4+2], S4=tot[c*4+3];
    const float m = S1 / N, ex2 = S2 / N, ex3 = S3 / N;
    const float var1 = (S2 - N * m * m) / (N - 1.f);
    const float sd = sqrtf(fmaxf(var1, 0.f));
    const float m3 = ex3 - 3.f*m*ex2 + 2.f*m*m*m;
    const float m4 = S4/N - 4.f*m*ex3 + 6.f*m*m*ex2 - 3.f*m*m*m*m;
    f[c*4+0]=m; f[c*4+1]=sd;
    f[c*4+2]=m3/(sd*sd*sd + 1e-6f);
    f[c*4+3]=m4/(sd*sd*sd*sd + 1e-6f);
  }
  const float my=tot[12]/N, mcb=tot[13]/N, mcr=tot[14]/N;
  const float inv = 1.f / (N - 1.f);
  f[12]=(tot[15]-N*my*my)  *inv; f[13]=(tot[16]-N*my*mcb) *inv;
  f[14]=(tot[17]-N*my*mcr) *inv; f[15]=(tot[18]-N*mcb*mcb)*inv;
  f[16]=(tot[19]-N*mcb*mcr)*inv; f[17]=(tot[20]-N*mcr*mcr)*inv;
  f[18]=mcb; f[19]=mcr;
  float ent = 0.f;
  const float denom = N + 1e-6f;
#pragma unroll
  for (int i = 0; i < 12; i++) {
    float pr = tot[32 + i] / denom;
    ent -= pr * logf(pr + 1e-6f);
  }
  f[35] = ent;
  const float ms = tot[21] / N;
  f[36] = ms;
  f[37] = sqrtf(fmaxf((tot[22] - N * ms * ms) / (N - 1.f), 0.f));
  const float mg0=tot[23]/N, mg1=tot[24]/N, mg2=tot[25]/N;
  const float c00=tot[26]-N*mg0*mg0, c11=tot[27]-N*mg1*mg1, c22=tot[28]-N*mg2*mg2;
  const float c01=tot[29]-N*mg0*mg1, c02=tot[30]-N*mg0*mg2, c12=tot[31]-N*mg1*mg2;
  const float d0=sqrtf(c00), d1=sqrtf(c11), d2=sqrtf(c22);
  const float r01=fminf(fmaxf(c01/(d0*d1),-1.f),1.f);
  const float r02=fminf(fmaxf(c02/(d0*d2),-1.f),1.f);
  const float r12=fminf(fmaxf(c12/(d1*d2),-1.f),1.f);
  f[38]=(r01+r02+r12)*(1.f/3.f);
}

// ---------------------------------------------------------------- per-(image,patch) stats + 8x8 bilinear
__global__ void k_patch_stats(const float* __restrict__ x,
                              float* __restrict__ arrs,   // 5 arrays of [128][16]
                              float* __restrict__ vout) { // [128][16][64]
  const int bp = blockIdx.x;
  const int b = bp >> 4, p = bp & 15;
  const int pi = p >> 2, pj = p & 3;
  const int tid = threadIdx.x;          // 256 threads
  const float* img = x + (size_t)b * 3 * HW;

  __shared__ float py[64 * 64];
  __shared__ float M8[8 * 64];
  __shared__ float t8[8 * 64];
  __shared__ float red[8 * 256];
  __shared__ float tot[8];

  if (tid < 8) {                         // bilinear matrix row (64 -> 8): 2-tap
    for (int i = 0; i < 64; i++) M8[tid * 64 + i] = 0.f;
    float src = (tid + 0.5f) * 8.f - 0.5f;
    src = fminf(fmaxf(src, 0.f), 63.f);
    int i0 = (int)floorf(src); i0 = i0 < 0 ? 0 : (i0 > 63 ? 63 : i0);
    int i1 = i0 + 1; if (i1 > 63) i1 = 63;
    float w = src - (float)i0;
    M8[tid * 64 + i0] += 1.f - w;
    M8[tid * 64 + i1] += w;
  }

  float a_y=0, a_yy=0, a_cb=0, a_cbcb=0, a_cr=0, a_crcr=0;
  for (int t = tid; t < 4096; t += 256) {
    const int i = t >> 6, j = t & 63;
    const int gp = (pi * 64 + i) * 256 + (pj * 64 + j);
    const float r = img[gp], g = img[HW + gp], bl = img[2 * HW + gp];
    const float y  =  0.299f    * r + 0.587f    * g + 0.114f    * bl;
    const float cb = -0.168736f * r - 0.331264f * g + 0.5f      * bl + 0.5f;
    const float cr =  0.5f      * r - 0.418688f * g - 0.081312f * bl + 0.5f;
    py[t] = y;
    a_y += y; a_yy += y*y; a_cb += cb; a_cbcb += cb*cb; a_cr += cr; a_crcr += cr*cr;
  }
  __syncthreads();

  float a_l=0, a_ll=0;                  // Laplacian, zero-padded per patch
  for (int t = tid; t < 4096; t += 256) {
    const int i = t >> 6, j = t & 63;
    const float c  = py[t];
    const float up = i > 0  ? py[t - 64] : 0.f;
    const float dn = i < 63 ? py[t + 64] : 0.f;
    const float lt = j > 0  ? py[t - 1]  : 0.f;
    const float rt = j < 63 ? py[t + 1]  : 0.f;
    const float lap = up + dn + lt + rt - 4.f * c;
    a_l += lap; a_ll += lap * lap;
  }
  red[0*256+tid]=a_y;  red[1*256+tid]=a_yy; red[2*256+tid]=a_cb; red[3*256+tid]=a_cbcb;
  red[4*256+tid]=a_cr; red[5*256+tid]=a_crcr; red[6*256+tid]=a_l; red[7*256+tid]=a_ll;
  __syncthreads();
  if (tid < 8) {
    float s = 0;
    for (int i = 0; i < 256; i++) s += red[tid * 256 + i];
    tot[tid] = s;
  }
  __syncthreads();
  if (tid == 0) {
    const float N = 4096.f, inv1 = 1.f / 4095.f;
    const float my = tot[0] / N;
    arrs[0*2048 + b*16 + p] = my;
    arrs[1*2048 + b*16 + p] = sqrtf(fmaxf((tot[1] - N*my*my) * inv1, 0.f));
    const float ml = tot[6] / N;
    arrs[2*2048 + b*16 + p] = (tot[7] - N*ml*ml) * inv1;
    const float mcb = tot[2] / N;
    arrs[3*2048 + b*16 + p] = sqrtf(fmaxf((tot[3] - N*mcb*mcb) * inv1, 0.f));
    const float mcr = tot[4] / N;
    arrs[4*2048 + b*16 + p] = sqrtf(fmaxf((tot[5] - N*mcr*mcr) * inv1, 0.f));
  }

  // t8 = M8 @ patch  (8x64), then v = t8 @ M8^T (8x8)
  for (int q = tid; q < 512; q += 256) {
    const int o = q >> 6, j = q & 63;
    float acc = 0.f;
    for (int i = 0; i < 64; i++) acc += M8[o * 64 + i] * py[i * 64 + j];
    t8[o * 64 + j] = acc;
  }
  __syncthreads();
  if (tid < 64) {
    const int o = tid >> 3, k = tid & 7;
    float acc = 0.f;
    for (int j = 0; j < 64; j++) acc += t8[o * 64 + j] * M8[k * 64 + j];
    vout[(size_t)(b * 16 + p) * 64 + o * 8 + k] = acc;
  }
}

// ---------------------------------------------------------------- per-image Gram via TDM + WMMA
__global__ void k_selfsim(const float* __restrict__ vin, const float* __restrict__ arrs,
                          float* __restrict__ fv) {
  const int b = blockIdx.x;
  const int lane = threadIdx.x;         // 32 threads (one wave)
  __shared__ float vv[16 * 64];
  __shared__ float sims[256];
  __shared__ float invn[16];
  const float* vb = vin + (size_t)b * 1024;

  // --- Tensor Data Mover: bulk copy 16x64 f32 tile (4KB, contiguous) global -> LDS.
  // D# per CDNA5 ISA 8.3/8.4: 1-D tile, data_size=4B, tile_dim0=tensor_dim0=1024.
  if (lane == 0) {
    const unsigned long long gaddr = (unsigned long long)(const void*)vb;
    const unsigned lds = (unsigned)(unsigned long long)(const void*)vv; // LDS offset = addr[31:0]
    u32x4 g0;
    g0.x = 1u;                                       // count=1, user mode, no gather
    g0.y = lds;                                      // lds_addr [63:32]
    g0.z = (unsigned)(gaddr & 0xFFFFFFFFull);        // global_addr [95:64]
    g0.w = (unsigned)((gaddr >> 32) & 0x1FFFFFFull)  // global_addr [120:96]
           | (2u << 30);                             // type=2 ("image") [127:126]
    i32x8 g1;
    g1[0] = 0x00020000;        // workgroup_mask=0, data_size=2 (4B)
    g1[1] = 0x04000000;        // tensor_dim0 low16 (1024) << 16
    g1[2] = 0x00010000;        // tensor_dim0 hi16=0 | tensor_dim1 low16 (1) << 16
    g1[3] = 0x04000000;        // tensor_dim1 hi16=0 | tile_dim0 (1024) << 16
    g1[4] = 0x00000001;        // tile_dim1=1, tile_dim2=0
    g1[5] = 1024;              // tensor_dim0_stride low32
    g1[6] = 0;                 // stride0 hi16=0, tensor_dim1_stride low16=0
    g1[7] = 0;
    i32x4 gz = {0, 0, 0, 0};   // groups 2/3 unused (<=2D tensor)
    i32x8 gz8 = {0, 0, 0, 0, 0, 0, 0, 0};
    __builtin_amdgcn_tensor_load_to_lds(g0, g1, gz, gz, gz8, 0);
    __builtin_amdgcn_s_wait_tensorcnt(0);
  }
  __syncthreads();

  if (lane < 16) {
    float n = 0.f;
    for (int k = 0; k < 64; k++) { float t = vv[lane * 64 + k]; n += t * t; }
    invn[lane] = 1.f / fmaxf(sqrtf(n), 1e-12f);
  }
  __syncthreads();
  for (int i = lane; i < 1024; i += 32) vv[i] *= invn[i >> 6];
  __syncthreads();

  const int row = lane & 15, half = lane >> 4;
  v8f c = {};
#pragma unroll
  for (int kh = 0; kh < 2; kh++) {      // K = 64 as 2 x (16x16x32)
    v16h a, bf;
#pragma unroll
    for (int h = 0; h < 16; h++) {      // ISA 16-bit A 16x32 per-lane layout
      const int g = h >> 1;
      const int kl = (g >> 2) * 16 + (g & 3) * 2 + half * 8 + (h & 1);
      const int k = kh * 32 + kl;
      a[h]  = (_Float16)vv[row * 64 + k];   // A[m][k], m = row
      bf[h] = (_Float16)vv[row * 64 + k];   // B[k][n] = V[n][k], n = row
    }
    c = __builtin_amdgcn_wmma_f32_16x16x32_f16(false, a, false, bf,
                                               (short)0, c, false, false);
  }
#pragma unroll
  for (int r = 0; r < 8; r++)
    sims[(r + half * 8) * 16 + row] = c[r];  // C/D: M = vgpr + 8*half, N = lane%16
  __syncthreads();

  if (lane == 0) {
    float sum=0, sq=0, mx=-2.f, cnt=0, t0=-2.f, t1=-2.f, t2=-2.f;
    for (int i = 0; i < 256; i++) {
      if ((i % 17) == 0) continue;      // skip diagonal
      const float s = sims[i];
      sum += s; sq += s * s; mx = fmaxf(mx, s);
      if (s > 0.9f) cnt += 1.f;
      if (s > t0) { t2 = t1; t1 = t0; t0 = s; }
      else if (s > t1) { t2 = t1; t1 = s; }
      else if (s > t2) { t2 = s; }
    }
    const float mean = sum / 240.f;
    const float sd = sqrtf(fmaxf((sq - 240.f * mean * mean) / 239.f, 0.f));
    float* f = fv + b * FVW;
    f[30]=mean; f[31]=sd; f[32]=mx; f[33]=cnt/240.f; f[34]=(t0+t1+t2)*(1.f/3.f);
  }
  if (lane < 5) {                       // mean/std over 16 patches, 5 arrays
    const float* a = arrs + lane * 2048 + b * 16;
    float s = 0, s2 = 0;
    for (int i = 0; i < 16; i++) { float t = a[i]; s += t; s2 += t * t; }
    const float m = s / 16.f;
    const float sd = sqrtf(fmaxf((s2 - 16.f * m * m) / 15.f, 0.f));
    float* f = fv + b * FVW;
    f[20 + lane * 2] = m; f[21 + lane * 2] = sd;
  }
}

// ---------------------------------------------------------------- MLP GEMMs via f32 WMMA
__global__ void k_gemm1(const float* __restrict__ fv, const float* __restrict__ W1,
                        float* __restrict__ z1) {     // (128x40) @ (40x128) -> 128x128
  const int tile = blockIdx.x;          // 64 tiles, 32 threads each
  const int mt = tile >> 3, nt = tile & 7;
  const int lane = threadIdx.x;
  const int r = lane & 15, half = lane >> 4;
  const int m = mt * 16 + r, n = nt * 16 + r;
  v8f c = {};
  for (int ks = 0; ks < 10; ks++) {     // K = 40 (col 39 zero-padded)
    v2f a, bf;
#pragma unroll
    for (int e = 0; e < 2; e++) {       // f32 A 16x4: k = e + 2*half
      const int k = ks * 4 + half * 2 + e;
      a[e]  = fv[m * FVW + k];
      bf[e] = (k < 39) ? W1[n * 39 + k] : 0.f;
    }
    c = __builtin_amdgcn_wmma_f32_16x16x4_f32(false, a, false, bf,
                                              (short)0, c, false, false);
  }
#pragma unroll
  for (int rr = 0; rr < 8; rr++)
    z1[(mt * 16 + rr + half * 8) * 128 + nt * 16 + r] = c[rr];
}

__global__ void k_gemm2(const float* __restrict__ z1ln, const float* __restrict__ W2,
                        float* __restrict__ z2) {     // (128x128) @ (128x32) -> 128x32
  const int tile = blockIdx.x;          // 16 tiles
  const int mt = tile >> 1, nt = tile & 1;
  const int lane = threadIdx.x;
  const int r = lane & 15, half = lane >> 4;
  const int m = mt * 16 + r, n = nt * 16 + r;
  v8f c = {};
  for (int ks = 0; ks < 32; ks++) {
    v2f a, bf;
#pragma unroll
    for (int e = 0; e < 2; e++) {
      const int k = ks * 4 + half * 2 + e;
      a[e]  = z1ln[m * 128 + k];
      bf[e] = W2[n * 128 + k];
    }
    c = __builtin_amdgcn_wmma_f32_16x16x4_f32(false, a, false, bf,
                                              (short)0, c, false, false);
  }
#pragma unroll
  for (int rr = 0; rr < 8; rr++)
    z2[(mt * 16 + rr + half * 8) * 32 + nt * 16 + r] = c[rr];
}

// ---------------------------------------------------------------- GELU + LayerNorm rows
__global__ void k_act1(const float* __restrict__ z1, const float* __restrict__ b1,
                       const float* __restrict__ g1, const float* __restrict__ be1,
                       float* __restrict__ z1ln) {
  const int b = blockIdx.x, t = threadIdx.x;   // 128 threads
  __shared__ float buf[128];
  __shared__ float rs[2];
  float v = z1[b * 128 + t] + b1[t];
  v = 0.5f * v * (1.f + erff(v * 0.70710678118654752f));
  buf[t] = v;
  __syncthreads();
  if (t == 0) {
    float s = 0; for (int i = 0; i < 128; i++) s += buf[i];
    const float m = s / 128.f;
    float q = 0; for (int i = 0; i < 128; i++) { float d = buf[i] - m; q += d * d; }
    rs[0] = m; rs[1] = q / 128.f;
  }
  __syncthreads();
  z1ln[b * 128 + t] = (v - rs[0]) * rsqrtf(rs[1] + 1e-5f) * g1[t] + be1[t];
}

__global__ void k_act2(const float* __restrict__ z2, const float* __restrict__ b2,
                       const float* __restrict__ g2, const float* __restrict__ be2,
                       float* __restrict__ out) {
  const int b = blockIdx.x, t = threadIdx.x;   // 32 threads
  __shared__ float buf[32];
  __shared__ float rs[2];
  float v = z2[b * 32 + t] + b2[t];
  v = 0.5f * v * (1.f + erff(v * 0.70710678118654752f));
  buf[t] = v;
  __syncthreads();
  if (t == 0) {
    float s = 0; for (int i = 0; i < 32; i++) s += buf[i];
    const float m = s / 32.f;
    float q = 0; for (int i = 0; i < 32; i++) { float d = buf[i] - m; q += d * d; }
    rs[0] = m; rs[1] = q / 32.f;
  }
  __syncthreads();
  out[b * 32 + t] = (v - rs[0]) * rsqrtf(rs[1] + 1e-5f) * g2[t] + be2[t];
}

// ---------------------------------------------------------------- launch
extern "C" void kernel_launch(void* const* d_in, const int* in_sizes, int n_in,
                              void* d_out, int out_size, void* d_ws, size_t ws_size,
                              hipStream_t stream) {
  const float* x   = (const float*)d_in[0];
  const float* W1  = (const float*)d_in[1];
  const float* b1  = (const float*)d_in[2];
  const float* g1  = (const float*)d_in[3];
  const float* be1 = (const float*)d_in[4];
  const float* W2  = (const float*)d_in[5];
  const float* b2  = (const float*)d_in[6];
  const float* g2  = (const float*)d_in[7];
  const float* be2 = (const float*)d_in[8];
  float* out = (float*)d_out;

  float* ws    = (float*)d_ws;
  float* fv    = ws;            // 128*40        = 5120
  float* arrs  = ws + 5120;     // 5*128*16      = 10240
  float* vbuf  = ws + 15360;    // 128*16*64     = 131072
  float* z1    = ws + 146432;   // 128*128       = 16384
  float* z1ln  = ws + 162816;   // 128*128       = 16384
  float* z2    = ws + 179200;   // 128*32        = 4096
  float* parts = ws + 183296;   // 1024*44       = 45056

  k_zero<<<(5120 + 255) / 256, 256, 0, stream>>>(fv, 5120);
  k_stats_part<<<128 * NSL, 256, 0, stream>>>(x, parts);
  k_stats_final<<<128, 64, 0, stream>>>(parts, fv);
  k_patch_stats<<<2048, 256, 0, stream>>>(x, arrs, vbuf);
  k_selfsim<<<128, 32, 0, stream>>>(vbuf, arrs, fv);
  k_gemm1<<<64, 32, 0, stream>>>(fv, W1, z1);
  k_act1<<<128, 128, 0, stream>>>(z1, b1, g1, be1, z1ln);
  k_gemm2<<<16, 32, 0, stream>>>(z1ln, W2, z2);
  k_act2<<<128, 32, 0, stream>>>(z2, b2, g2, be2, out);
}
